// LocalRelationalLayer_37795712205032
// MI455X (gfx1250) — compile-verified, matrix operand order checked by reading, not run on hardware
//
#include <hip/hip_runtime.h>
#include <hip/hip_bf16.h>
#include <math.h>

typedef __attribute__((ext_vector_type(16))) _Float16 v16h;
typedef __attribute__((ext_vector_type(8)))  float    v8f;

#define B_  8
#define C_  128
#define C1_ 16
#define H_  56
#define W_  56
#define L_  3136          // 56*56
#define K_  7
#define K2_ 49
#define MID_ 24
#define M_  8
#define NTILE_ ((B_ * L_) / 16)   // 1568 16-pixel tiles

// A/B-fragment K index for v16h element (v, e) and lane group (ISA 7.12.2,
// 16-bit 16x32): lanes 0-15 -> K{0..7}U{16..23}; lanes 16-31 -> K{8..15}U{24..31}.
__device__ __forceinline__ int frag_k(int v, int e, int grp) {
    int base = (v < 4) ? (2 * v) : (16 + 2 * (v - 4));
    return base + e + grp * 8;
}

// ---------------------------------------------------------------------------
// Prep: swizzle kw/qw/fw (f32, row-major (O,C)) into f16 WMMA B-fragment
// layout so GEMM waves load one contiguous v16h per fragment.
//   fwf: [step(4)][og(8)][lane(32)][h(16)]   kwf/qwf: [step(4)][lane(32)][h(16)]
// ---------------------------------------------------------------------------
__global__ void prep_kernel(const float* __restrict__ kw, const float* __restrict__ qw,
                            const float* __restrict__ fw,
                            _Float16* __restrict__ kwf, _Float16* __restrict__ qwf,
                            _Float16* __restrict__ fwf)
{
    const int t = blockIdx.x * blockDim.x + threadIdx.x;
    if (t >= 1280) return;
    _Float16 tmp[16];
    if (t < 1024) {                               // fw fragments
        const int lane = t & 31, og = (t >> 5) & 7, step = t >> 8;
        const int n = lane & 15, grp = lane >> 4;
        #pragma unroll
        for (int v = 0; v < 8; ++v)
            #pragma unroll
            for (int e = 0; e < 2; ++e)
                tmp[2 * v + e] = (_Float16)fw[(og * 16 + n) * C_ + step * 32 + frag_k(v, e, grp)];
        _Float16* dst = fwf + (((size_t)step * 8 + og) * 32 + lane) * 16;
        #pragma unroll
        for (int h = 0; h < 16; ++h) dst[h] = tmp[h];
    } else {                                      // kw (128 threads) then qw (128)
        const int u = t - 1024;
        const int which = u >> 7;
        const int uu = u & 127;
        const int lane = uu & 31, step = uu >> 5;
        const int n = lane & 15, grp = lane >> 4;
        const float* src = which ? qw : kw;
        _Float16* dst = (which ? qwf : kwf) + ((size_t)step * 32 + lane) * 16;
        #pragma unroll
        for (int v = 0; v < 8; ++v)
            #pragma unroll
            for (int e = 0; e < 2; ++e)
                tmp[2 * v + e] = (_Float16)src[n * C_ + step * 32 + frag_k(v, e, grp)];
        #pragma unroll
        for (int h = 0; h < 16; ++h) dst[h] = tmp[h];
    }
}

// ---------------------------------------------------------------------------
// Geometry prior g[c1][ki*7+kj] (tiny MLP on pos).
// ---------------------------------------------------------------------------
__global__ void geom_kernel(const float* __restrict__ pos,
                            const float* __restrict__ g1w, const float* __restrict__ g1b,
                            const float* __restrict__ g2w, const float* __restrict__ g2b,
                            float* __restrict__ g)
{
    const int t = blockIdx.x * blockDim.x + threadIdx.x;
    if (t >= C1_ * K2_) return;
    const int c1 = t / K2_, j = t % K2_;
    float acc = g2b[c1];
    #pragma unroll
    for (int h = 0; h < 8; ++h) {
        float u = g1b[h] + g1w[h * 2 + 0] * pos[0 * K2_ + j]
                         + g1w[h * 2 + 1] * pos[1 * K2_ + j];
        u = fmaxf(u, 0.f);
        acc += g2w[c1 * 8 + h] * u;
    }
    g[t] = acc;
}

// ---------------------------------------------------------------------------
// Kernel A: km/qm = conv1x1(x, kw/qw) via WMMA.  One wave per 16-pixel tile;
// A loaded (coalesced per-channel rows) from x; B fragments are single v16h
// loads from the pre-swizzled weight buffers.  K loop fully unrolled.
// ---------------------------------------------------------------------------
__global__ void kq_gemm_kernel(const float* __restrict__ x,
                               const _Float16* __restrict__ kwf, const float* __restrict__ kbias,
                               const _Float16* __restrict__ qwf, const float* __restrict__ qbias,
                               float* __restrict__ km, float* __restrict__ qm)
{
    const int lane = threadIdx.x & 31;
    const int wid  = blockIdx.x * (blockDim.x >> 5) + (threadIdx.x >> 5);
    if (wid >= NTILE_) return;                    // wave-uniform

    const int p0  = wid * 16;                     // tiles never straddle batches
    const int b   = p0 / L_;
    const int s0  = p0 % L_;
    const int m   = lane & 15;
    const int grp = lane >> 4;
    const int n   = lane & 15;

    const float* xb  = x + (size_t)b * C_ * L_;
    const v16h*  kwv = (const v16h*)kwf;
    const v16h*  qwv = (const v16h*)qwf;

    v8f acck = {}, accq = {};
    #pragma unroll
    for (int step = 0; step < 4; ++step) {
        v16h a;
        #pragma unroll
        for (int v = 0; v < 8; ++v) {
            #pragma unroll
            for (int e = 0; e < 2; ++e) {
                const int c = step * 32 + frag_k(v, e, grp);
                a[2 * v + e] = (_Float16)xb[(size_t)c * L_ + s0 + m];
            }
        }
        const v16h bk = kwv[step * 32 + lane];
        const v16h bq = qwv[step * 32 + lane];
        acck = __builtin_amdgcn_wmma_f32_16x16x32_f16(false, a, false, bk, (short)0, acck, false, false);
        accq = __builtin_amdgcn_wmma_f32_16x16x32_f16(false, a, false, bq, (short)0, accq, false, false);
    }

    #pragma unroll
    for (int r = 0; r < 8; ++r) {
        const int prow = s0 + r + 8 * grp;        // pixel row M = r + 8*grp
        km[((size_t)b * C1_ + n) * L_ + prow] = acck[r] + kbias[n];
        qm[((size_t)b * C1_ + n) * L_ + prow] = accq[r] + qbias[n];
    }
}

// ---------------------------------------------------------------------------
// Kernel C: ak = ku*qu[mid] + g, softmax over last-K axis, weighted sum of
// unfolded x.  Faithful to torch's raw .view(): (c1,l,j) -> t = l*49+j,
// kp = t/L, sp = t%L over the (C1*49, L) unfold buffer.
// Writes pre directly in f16 WMMA A-fragment order:
//   preh: [tile][step(4)][lane(32)][h(16)]
// Inverse fragment map for pixel lm=l%16, channel ch=mm*16+c1:
//   step = ch/32, kk = 16*(mm&1)+c1, grp = (c1>=8), lane = lm+16*grp,
//   h = 8*(mm&1) + (c1&7).
// ---------------------------------------------------------------------------
__global__ void attn_kernel(const float* __restrict__ x,
                            const float* __restrict__ km, const float* __restrict__ qm,
                            const float* __restrict__ g,  _Float16* __restrict__ preh)
{
    const int t = blockIdx.x * blockDim.x + threadIdx.x;
    if (t >= B_ * C1_ * L_) return;
    const int l  = t % L_;
    const int c1 = (t / L_) % C1_;
    const int b  = t / (C1_ * L_);

    const float* kmb = km + ((size_t)b * C1_ + c1) * L_;
    const float* qmb = qm + ((size_t)b * C1_ + c1) * L_;
    const float* xb  = x  + (size_t)b * C_ * L_;

    int   off[K2_];
    float ck[K2_];
    #pragma unroll
    for (int j = 0; j < K2_; ++j) {
        const int tt = l * K2_ + j;
        const int kp = tt / L_;
        const int sp = tt - kp * L_;
        const int ki = kp / K_;
        const int kj = kp - ki * K_;
        const int sh = sp / W_;
        const int sw = sp - sh * W_;
        const int hh = ki + sh - (K_ / 2);
        const int ww = kj + sw - (K_ / 2);
        const bool ok = ((unsigned)hh < (unsigned)H_) && ((unsigned)ww < (unsigned)W_);
        off[j] = ok ? (hh * W_ + ww) : -1;
    }

    const float qmid = (off[MID_] >= 0) ? qmb[off[MID_]] : 0.f;

    #pragma unroll
    for (int j = 0; j < K2_; ++j) {
        const float kv = (off[j] >= 0) ? kmb[off[j]] : 0.f;
        ck[j] = kv * qmid + g[c1 * K2_ + j];
    }

    // Softmax over kj (rows of 7) only.
    #pragma unroll
    for (int ki = 0; ki < K_; ++ki) {
        float mx = ck[ki * K_];
        #pragma unroll
        for (int kj = 1; kj < K_; ++kj) mx = fmaxf(mx, ck[ki * K_ + kj]);
        float sum = 0.f;
        #pragma unroll
        for (int kj = 0; kj < K_; ++kj) {
            const float e = __expf(ck[ki * K_ + kj] - mx);
            ck[ki * K_ + kj] = e;
            sum += e;
        }
        const float inv = 1.f / sum;
        #pragma unroll
        for (int kj = 0; kj < K_; ++kj) ck[ki * K_ + kj] *= inv;
    }

    // Mask OOB contributions; clamp offsets so gather loads are branch-free.
    #pragma unroll
    for (int j = 0; j < K2_; ++j) {
        if (off[j] < 0) { ck[j] = 0.f; off[j] = 0; }
    }

    const int tile  = (b * L_ + l) >> 4;
    const int lm    = l & 15;
    const int grp2  = (c1 >> 3) & 1;
    const int lane2 = lm + 16 * grp2;
    const int hbase = c1 & 7;

    #pragma unroll
    for (int mm = 0; mm < M_; ++mm) {
        const int ch = mm * C1_ + c1;
        const float* xc = xb + (size_t)ch * L_;
        float acc = 0.f;
        #pragma unroll
        for (int j = 0; j < K2_; ++j) acc += ck[j] * xc[off[j]];

        const int step = mm >> 1;
        const int h    = 8 * (mm & 1) + hbase;
        preh[(((size_t)tile * 4 + step) * 32 + lane2) * 16 + h] = (_Float16)acc;
    }
}

// ---------------------------------------------------------------------------
// Kernel D: out = conv1x1(pre, fw) + fb via WMMA.  One wave: 16 pixels x 64
// outputs.  A and B fragments are single contiguous v16h loads (pre-swizzled),
// so the inner loop is b128 loads + wmma only.  K loop fully unrolled.
// ---------------------------------------------------------------------------
__global__ void out_gemm_kernel(const _Float16* __restrict__ preh,
                                const _Float16* __restrict__ fwf, const float* __restrict__ fb,
                                float* __restrict__ out)
{
    const int lane = threadIdx.x & 31;
    const int wid  = blockIdx.x * (blockDim.x >> 5) + (threadIdx.x >> 5);
    if (wid >= NTILE_ * 2) return;                // wave-uniform

    const int tile  = wid >> 1;
    const int ohalf = wid & 1;
    const int p0 = tile * 16;
    const int b  = p0 / L_;
    const int s0 = p0 % L_;
    const int grp = lane >> 4;
    const int n   = lane & 15;

    const v16h* pf = (const v16h*)preh;
    const v16h* wf = (const v16h*)fwf;

    v8f acc[4] = {};
    #pragma unroll
    for (int step = 0; step < 4; ++step) {
        const v16h a = pf[((size_t)tile * 4 + step) * 32 + lane];
        #pragma unroll
        for (int nn = 0; nn < 4; ++nn) {
            const v16h bf = wf[((size_t)step * 8 + ohalf * 4 + nn) * 32 + lane];
            acc[nn] = __builtin_amdgcn_wmma_f32_16x16x32_f16(false, a, false, bf, (short)0, acc[nn], false, false);
        }
    }

    #pragma unroll
    for (int nn = 0; nn < 4; ++nn) {
        const int o = ohalf * 64 + nn * 16 + n;
        #pragma unroll
        for (int r = 0; r < 8; ++r) {
            const int prow = s0 + r + 8 * grp;
            out[((size_t)b * C_ + o) * L_ + prow] = acc[nn][r] + fb[o];
        }
    }
}

// ---------------------------------------------------------------------------
// Launch
// ---------------------------------------------------------------------------
extern "C" void kernel_launch(void* const* d_in, const int* in_sizes, int n_in,
                              void* d_out, int out_size, void* d_ws, size_t ws_size,
                              hipStream_t stream) {
    (void)in_sizes; (void)n_in; (void)out_size; (void)ws_size;

    const float* x   = (const float*)d_in[0];
    const float* pos = (const float*)d_in[1];
    const float* kw  = (const float*)d_in[2];
    const float* kb  = (const float*)d_in[3];
    const float* qw  = (const float*)d_in[4];
    const float* qb  = (const float*)d_in[5];
    const float* g1w = (const float*)d_in[6];
    const float* g1b = (const float*)d_in[7];
    const float* g2w = (const float*)d_in[8];
    const float* g2b = (const float*)d_in[9];
    const float* fw  = (const float*)d_in[10];
    const float* fb  = (const float*)d_in[11];
    float* out = (float*)d_out;

    // Workspace layout (float offsets, all 32B-aligned):
    //   km   @ 0        : 401408 f32
    //   qm   @ 401408   : 401408 f32
    //   g    @ 802816   : 784 f32 (padded to 803840)
    //   preh @ 803840   : 3211264 f16 (= 1605632 f32 slots), A-fragment order
    //   fwf  @ 2409472  : 16384 f16 (8192 f32 slots), B-fragment order
    //   kwf  @ 2417664  : 2048 f16 (1024 f32 slots)
    //   qwf  @ 2418688  : 2048 f16 (1024 f32 slots)      total ~9.7 MB
    float*     ws   = (float*)d_ws;
    float*     km   = ws;
    float*     qm   = ws + 401408;
    float*     g    = ws + 802816;
    _Float16*  preh = (_Float16*)(ws + 803840);
    _Float16*  fwf  = (_Float16*)(ws + 2409472);
    _Float16*  kwf  = (_Float16*)(ws + 2417664);
    _Float16*  qwf  = (_Float16*)(ws + 2418688);

    // Stage 0: weight swizzle + geometry prior (tiny).
    prep_kernel<<<5, 256, 0, stream>>>(kw, qw, fw, kwf, qwf, fwf);
    geom_kernel<<<4, 256, 0, stream>>>(pos, g1w, g1b, g2w, g2b, g);

    // Stage 1: km/qm GEMM — 1568 waves, 4 waves (128 threads) per block.
    kq_gemm_kernel<<<392, 128, 0, stream>>>(x, kwf, kb, qwf, qb, km, qm);

    // Stage 2: attention — one thread per (b,c1,l) = 401408 threads.
    attn_kernel<<<(B_ * C1_ * L_ + 255) / 256, 256, 0, stream>>>(x, km, qm, g, preh);

    // Stage 3: final 1x1 conv — 3136 waves, 4 per block.
    out_gemm_kernel<<<784, 128, 0, stream>>>(preh, fwf, fb, out);
}
